// RelationAwareTransformerLayer_59244778881417
// MI455X (gfx1250) — matrix operand order repaired and checked by reading, not compiled, not run on hardware
//
#include <hip/hip_runtime.h>
#include <hip/hip_bf16.h>
#include <cstdint>
#include <cstddef>

// ---------------- problem constants ----------------
constexpr int Dm  = 1024;
constexpr int Hn  = 16;
constexpr int DHn = 64;
constexpr int FFn = 4096;
constexpr int Sn  = 2048;
constexpr int Bn  = 2;
constexpr int Mn  = Bn * Sn;   // 4096 rows

typedef __attribute__((ext_vector_type(16))) __bf16 v16bf;
typedef __attribute__((ext_vector_type(8)))  float  v8f;
typedef __attribute__((ext_vector_type(4)))  unsigned int u32x4;
typedef __attribute__((ext_vector_type(8)))  int i32x8;
typedef __attribute__((ext_vector_type(4)))  int i32x4;

// Tensor Data Mover available? (device pass only; host pass always takes fallback)
#if defined(__AMDGCN__) && __has_builtin(__builtin_amdgcn_tensor_load_to_lds) && __has_builtin(__builtin_amdgcn_s_wait_tensorcnt)
#define ATHENA_TDM 1
#else
#define ATHENA_TDM 0
#endif

// ---------------- WMMA helpers (CDNA5 wave32) ----------------
__device__ inline v8f wmma_bf16(v16bf a, v16bf b, v8f c) {
  // D = A(16x32) * B(32x16) + C(16x16 f32)
  return __builtin_amdgcn_wmma_f32_16x16x32_bf16(
      /*neg_a=*/false, a, /*neg_b=*/false, b,
      /*c_mod=*/(short)0, c, /*reuse_a=*/false, /*reuse_b=*/false);
}

// A-matrix fragment, 16x32 bf16, K contiguous. Lane r=lane&15 -> row M;
// half h: element j -> K=16h+j (matches load_frag_b).
__device__ inline v16bf load_frag_a(const __bf16* __restrict__ p, int ldr) {
  const int lane = threadIdx.x & 31;
  const int r = lane & 15, h = lane >> 4;
  const __bf16* row = p + (size_t)r * ldr;
  v16bf f;
#pragma unroll
  for (int j = 0; j < 16; ++j) f[j] = row[16 * h + j];
  return f;
}

// B-matrix fragment, 32x16 bf16. Lane r -> column N; half h covers K=16h..16h+15.
__device__ inline v16bf load_frag_b(const __bf16* __restrict__ p, int ldr) {
  const int lane = threadIdx.x & 31;
  const int r = lane & 15, h = lane >> 4;
  const __bf16* row = p + (size_t)r * ldr + 16 * h;
  v16bf f;
#pragma unroll
  for (int j = 0; j < 16; ++j) f[j] = row[j];
  return f;
}

// B fragment with arbitrary strides (ldr along N, ldk along K) — V tiles in LDS.
__device__ inline v16bf load_frag_b_strided(const __bf16* __restrict__ p, long ldr, long ldk) {
  const int lane = threadIdx.x & 31;
  const int r = lane & 15, h = lane >> 4;
  v16bf f;
#pragma unroll
  for (int j = 0; j < 16; ++j) f[j] = p[(long)r * ldr + (long)(16 * h + j) * ldk];
  return f;
}

#if ATHENA_TDM
__device__ inline unsigned lds_off(const void* p) {
  // generic pointer to LDS: low 32 bits are the LDS byte offset (flat aperture mapping)
  return (unsigned)(uintptr_t)p;
}

// Post a 2-D TDM tile load: tile_h rows x tile_w bf16 elems, row stride row_stride
// elems, from global `g` into LDS offset `lo`. Descriptor per ISA §8.3/8.4:
// group0: count=1 | lds_addr | global_addr[56:0] | type=2
// group1: data_size=1(2B), tensor_dim0=row_stride, tensor_dim1=tile_h,
//         tile_dim0=tile_w, tile_dim1=tile_h, tensor_dim0_stride=row_stride.
__device__ inline void tdm_load_2d(unsigned lo, const void* g,
                                   unsigned tile_w, unsigned tile_h,
                                   unsigned row_stride) {
  const unsigned long long ga = (unsigned long long)(uintptr_t)g;
  const unsigned tw = row_stride;   // tensor dim0 (elems) — no clipping inside tile
  const unsigned th = tile_h;
  u32x4 g0;
  g0[0] = 1u;                                           // count=1, user mode
  g0[1] = lo;                                           // lds_addr (bytes)
  g0[2] = (unsigned)(ga & 0xffffffffu);                 // global_addr[31:0]
  g0[3] = (unsigned)((ga >> 32) & 0x01ffffffu) | (2u << 30);  // addr[56:32] | type=2
  i32x8 g1;
  g1[0] = (int)(1u << 16);                              // data_size=1 (2 bytes)
  g1[1] = (int)((tw & 0xffffu) << 16);                  // tensor_dim0[15:0]
  g1[2] = (int)(((tw >> 16) & 0xffffu) | ((th & 0xffffu) << 16));  // dim0[31:16]|dim1[15:0]
  g1[3] = (int)(((th >> 16) & 0xffffu) | ((tile_w & 0xffffu) << 16)); // dim1[31:16]|tile_dim0
  g1[4] = (int)(tile_h & 0xffffu);                      // tile_dim1 (tile_dim2=0)
  g1[5] = (int)row_stride;                              // tensor_dim0_stride[31:0]
  g1[6] = 0;                                            // stride[47:32]=0, dim1_stride lo=0
  g1[7] = 0;
  const i32x4 z4 = {0, 0, 0, 0};
#if __clang_major__ >= 23
  const i32x8 z8 = {0, 0, 0, 0, 0, 0, 0, 0};
  __builtin_amdgcn_tensor_load_to_lds(g0, g1, z4, z4, z8, 0);
#else
  __builtin_amdgcn_tensor_load_to_lds(g0, g1, z4, z4, 0);
#endif
}
#endif  // ATHENA_TDM

// ---------------- weight convert + transpose: W[K,N] f32 -> Wt[N,K] bf16 ----------------
__global__ void cvt_transpose_kernel(const float* __restrict__ W, __bf16* __restrict__ Wt,
                                     int K, int N) {
  size_t idx = (size_t)blockIdx.x * 256 + threadIdx.x;
  size_t total = (size_t)K * N;
  if (idx >= total) return;
  int k = (int)(idx % (size_t)K);
  size_t n = idx / (size_t)K;
  Wt[idx] = (__bf16)W[(size_t)k * N + n];
}

// ---------------- LayerNorm -> bf16 ----------------
__global__ __launch_bounds__(256) void ln_bf16_kernel(const float* __restrict__ x,
                                                      const float* __restrict__ g,
                                                      const float* __restrict__ b,
                                                      __bf16* __restrict__ out, int ncols) {
  const int row = blockIdx.x;
  const float* xr = x + (size_t)row * ncols;
  float s = 0.f, s2 = 0.f;
  for (int i = threadIdx.x; i < ncols; i += 256) {
    float v = xr[i];
    s += v; s2 += v * v;
  }
  __shared__ float red0[256], red1[256];
  red0[threadIdx.x] = s; red1[threadIdx.x] = s2;
  __syncthreads();
  for (int st = 128; st > 0; st >>= 1) {
    if (threadIdx.x < st) {
      red0[threadIdx.x] += red0[threadIdx.x + st];
      red1[threadIdx.x] += red1[threadIdx.x + st];
    }
    __syncthreads();
  }
  const float mean = red0[0] / ncols;
  const float var  = red1[0] / ncols - mean * mean;
  const float rstd = rsqrtf(var + 1e-5f);
  for (int i = threadIdx.x; i < ncols; i += 256)
    out[(size_t)row * ncols + i] = (__bf16)((xr[i] - mean) * rstd * g[i] + b[i]);
}

// ---------------- bf16 WMMA GEMM: C[M,N] = act(A[M,K] * Bt[N,K]^T + bias) (+residual) ----
// 256 threads = 8 waves; block tile 128x64; wave tile 32x32; K-step 32.
// Fast path: TDM fills double-buffered LDS tiles; waves only wait on TENSORcnt.
__global__ __launch_bounds__(256) void gemm_bf16_kernel(
    const __bf16* __restrict__ A, const __bf16* __restrict__ Bt,
    const float* __restrict__ bias, const float* __restrict__ residual,
    float* __restrict__ outf, __bf16* __restrict__ outb,
    int M, int N, int K, int do_gelu) {
  __shared__ __attribute__((aligned(16))) __bf16 As[2][128 * 32];
  __shared__ __attribute__((aligned(16))) __bf16 Bs[2][64 * 32];

  const int t = threadIdx.x;
  const int lane = t & 31, wave = t >> 5;
  const int wm = wave >> 1, wn = wave & 1;
  const int bm0 = blockIdx.y * 128, bn0 = blockIdx.x * 64;
  const int nk = K >> 5;

  v8f acc00 = {}, acc01 = {}, acc10 = {}, acc11 = {};

#if ATHENA_TDM
  if (wave == 0) {
    tdm_load_2d(lds_off(&As[0][0]), A + (size_t)bm0 * K, 32, 128, K);
    tdm_load_2d(lds_off(&Bs[0][0]), Bt + (size_t)bn0 * K, 32, 64, K);
  }
  for (int kt = 0; kt < nk; ++kt) {
    const int buf = kt & 1;
    if (wave == 0) {
      if (kt + 1 < nk) {
        tdm_load_2d(lds_off(&As[buf ^ 1][0]), A + (size_t)bm0 * K + (kt + 1) * 32, 32, 128, K);
        tdm_load_2d(lds_off(&Bs[buf ^ 1][0]), Bt + (size_t)bn0 * K + (kt + 1) * 32, 32, 64, K);
        __builtin_amdgcn_s_wait_tensorcnt(2);   // current buf's pair complete
      } else {
        __builtin_amdgcn_s_wait_tensorcnt(0);
      }
    }
    __syncthreads();
    const __bf16* Ab = &As[buf][0];
    const __bf16* Bb = &Bs[buf][0];
    v16bf a0 = load_frag_a(Ab + (wm * 32 + 0) * 32, 32);
    v16bf a1 = load_frag_a(Ab + (wm * 32 + 16) * 32, 32);
    v16bf b0 = load_frag_b(Bb + (wn * 32 + 0) * 32, 32);
    v16bf b1 = load_frag_b(Bb + (wn * 32 + 16) * 32, 32);
    acc00 = wmma_bf16(a0, b0, acc00);
    acc01 = wmma_bf16(a0, b1, acc01);
    acc10 = wmma_bf16(a1, b0, acc10);
    acc11 = wmma_bf16(a1, b1, acc11);
    __syncthreads();    // reads done before TDM overwrites this buffer next round
  }
#else
  const int ar = t >> 1, ac = (t & 1) * 16;
  const int br = t >> 2, bcx = (t & 3) * 8;
  for (int kt = 0; kt < nk; ++kt) {
    const int k0 = kt * 32;
    const __bf16* ag = A + (size_t)(bm0 + ar) * K + k0 + ac;
    const __bf16* bg = Bt + (size_t)(bn0 + br) * K + k0 + bcx;
    int4 av0 = *(const int4*)(ag);
    int4 av1 = *(const int4*)(ag + 8);
    int4 bv0 = *(const int4*)(bg);
    __builtin_prefetch((const void*)(ag + 32), 0, 1);
    __builtin_prefetch((const void*)(bg + 32), 0, 1);
    __syncthreads();
    *(int4*)(&As[0][0] + ar * 32 + ac) = av0;
    *(int4*)(&As[0][0] + ar * 32 + ac + 8) = av1;
    *(int4*)(&Bs[0][0] + br * 32 + bcx) = bv0;
    __syncthreads();
    v16bf a0 = load_frag_a(&As[0][0] + (wm * 32 + 0) * 32, 32);
    v16bf a1 = load_frag_a(&As[0][0] + (wm * 32 + 16) * 32, 32);
    v16bf b0 = load_frag_b(&Bs[0][0] + (wn * 32 + 0) * 32, 32);
    v16bf b1 = load_frag_b(&Bs[0][0] + (wn * 32 + 16) * 32, 32);
    acc00 = wmma_bf16(a0, b0, acc00);
    acc01 = wmma_bf16(a0, b1, acc01);
    acc10 = wmma_bf16(a1, b0, acc10);
    acc11 = wmma_bf16(a1, b1, acc11);
  }
#endif

  const int rr = lane & 15, hh = lane >> 4;
#pragma unroll
  for (int y = 0; y < 2; ++y) {
#pragma unroll
    for (int x = 0; x < 2; ++x) {
      const v8f accv = (y == 0) ? (x == 0 ? acc00 : acc01) : (x == 0 ? acc10 : acc11);
      const int n = bn0 + wn * 32 + x * 16 + rr;
      const float bn = bias ? bias[n] : 0.f;
#pragma unroll
      for (int i = 0; i < 8; ++i) {
        const int m = bm0 + wm * 32 + y * 16 + i + 8 * hh;
        float v = accv[i] + bn;
        if (do_gelu) v = 0.5f * v * (1.f + erff(v * 0.70710678118f));
        if (residual) v += residual[(size_t)m * N + n];
        if (outf) outf[(size_t)m * N + n] = v;
        if (outb) outb[(size_t)m * N + n] = (__bf16)v;
      }
    }
  }
}

// ---------------- flash attention with structural + metadata bias ----------------
// grid: (S/128, H, B); 8 waves; each wave owns 16 queries. The 32-key x 64-dh K and V
// tiles are staged into LDS ONCE PER BLOCK (TDM double-buffered on fast path),
// shared by all 8 waves.
__global__ __launch_bounds__(256) void attention_kernel(
    const __bf16* __restrict__ q, const __bf16* __restrict__ k, const __bf16* __restrict__ v,
    const float* __restrict__ abias, const float* __restrict__ meta,
    const unsigned char* __restrict__ kmask,
    const float* __restrict__ Ws, const float* __restrict__ bsv,
    const float* __restrict__ Wm, const float* __restrict__ bmv,
    __bf16* __restrict__ attn) {
  __shared__ __attribute__((aligned(16))) __bf16 Klds[2][32][64];
  __shared__ __attribute__((aligned(16))) __bf16 Vlds[2][32][64];
  __shared__ __attribute__((aligned(16))) __bf16 plds[8][16][32];

  const int t = threadIdx.x, lane = t & 31, w = t >> 5;
  const int r = lane & 15, hh = lane >> 4;
  const int b = blockIdx.z, h = blockIdx.y;
  const int i0 = blockIdx.x * 128 + w * 16;
  const float scale = 0.125f;  // DH^-0.5, DH=64

  const float ws0 = Ws[h], ws1 = Ws[Hn + h], ws2 = Ws[2 * Hn + h];
  const float wmh = Wm[h];
  const float bconst = bsv[h] + bmv[h];

  const __bf16* qbase = q + ((size_t)(b * Sn + i0) * Hn + h) * DHn;
  const v16bf aq0 = load_frag_a(qbase, Hn * DHn);        // dh 0..31
  const v16bf aq1 = load_frag_a(qbase + 32, Hn * DHn);   // dh 32..63

  v8f o0 = {}, o1 = {}, o2 = {}, o3 = {};
  float mrun[8], lrun[8];
#pragma unroll
  for (int i = 0; i < 8; ++i) { mrun[i] = -1e30f; lrun[i] = 0.f; }

  const __bf16* kg0 = k + ((size_t)(b * Sn) * Hn + h) * DHn;
  const __bf16* vg0 = v + ((size_t)(b * Sn) * Hn + h) * DHn;
  const int njt = Sn / 32;

#if ATHENA_TDM
  if (w == 0) {
    tdm_load_2d(lds_off(&Klds[0][0][0]), kg0, 64, 32, Hn * DHn);
    tdm_load_2d(lds_off(&Vlds[0][0][0]), vg0, 64, 32, Hn * DHn);
  }
#endif

  for (int jt = 0; jt < njt; ++jt) {
    const int j0 = jt * 32;
#if ATHENA_TDM
    const int buf = jt & 1;
    if (w == 0) {
      if (jt + 1 < njt) {
        tdm_load_2d(lds_off(&Klds[buf ^ 1][0][0]), kg0 + (size_t)(j0 + 32) * Hn * DHn, 64, 32, Hn * DHn);
        tdm_load_2d(lds_off(&Vlds[buf ^ 1][0][0]), vg0 + (size_t)(j0 + 32) * Hn * DHn, 64, 32, Hn * DHn);
        __builtin_amdgcn_s_wait_tensorcnt(2);
      } else {
        __builtin_amdgcn_s_wait_tensorcnt(0);
      }
    }
    __syncthreads();
#else
    const int buf = 0;
    {  // cooperative staging: 8 elems/thread per tile, b128 in and out
      const int row = t >> 3, seg = (t & 7) * 8;
      const __bf16* kgp = kg0 + (size_t)(j0 + row) * Hn * DHn + seg;
      const __bf16* vgp = vg0 + (size_t)(j0 + row) * Hn * DHn + seg;
      int4 kv = *(const int4*)kgp;
      int4 vv = *(const int4*)vgp;
      *(int4*)&Klds[0][row][seg] = kv;
      *(int4*)&Vlds[0][row][seg] = vv;
    }
    __syncthreads();
#endif

    // ---- S = Q K^T for 16 queries x 32 keys (two 16x16 C frags) ----
    v16bf bk00 = load_frag_b(&Klds[buf][0][0], 64);
    v16bf bk01 = load_frag_b(&Klds[buf][0][0] + 32, 64);
    v16bf bk10 = load_frag_b(&Klds[buf][16][0], 64);
    v16bf bk11 = load_frag_b(&Klds[buf][16][0] + 32, 64);
    v8f s0 = {}, s1 = {};
    s0 = wmma_bf16(aq0, bk00, s0);
    s0 = wmma_bf16(aq1, bk01, s0);
    s1 = wmma_bf16(aq0, bk10, s1);
    s1 = wmma_bf16(aq1, bk11, s1);

    const int gj0 = j0 + r, gj1 = j0 + 16 + r;
    const bool msk0 = kmask[(size_t)b * Sn + gj0] != 0;
    const bool msk1 = kmask[(size_t)b * Sn + gj1] != 0;

    // ---- bias add + online softmax (row m = i + 8*hh lives in one 16-lane half) ----
#pragma unroll
    for (int i = 0; i < 8; ++i) {
      const int gi = i0 + i + 8 * hh;
      const size_t base0 = ((size_t)b * Sn + gi) * Sn + gj0;
      const size_t base1 = ((size_t)b * Sn + gi) * Sn + gj1;
      const float* a0 = abias + base0 * 3;
      const float* a1 = abias + base1 * 3;
      float v0 = s0[i] * scale + a0[0] * ws0 + a0[1] * ws1 + a0[2] * ws2 + meta[base0] * wmh + bconst;
      float v1 = s1[i] * scale + a1[0] * ws0 + a1[1] * ws1 + a1[2] * ws2 + meta[base1] * wmh + bconst;
      if (msk0) v0 = -1e9f;
      if (msk1) v1 = -1e9f;

      float mx = fmaxf(v0, v1);
#pragma unroll
      for (int d = 1; d < 16; d <<= 1) mx = fmaxf(mx, __shfl_xor(mx, d, 32));
      const float mn = fmaxf(mrun[i], mx);
      const float alpha = __expf(mrun[i] - mn);
      v0 = __expf(v0 - mn);
      v1 = __expf(v1 - mn);
      float rs = v0 + v1;
#pragma unroll
      for (int d = 1; d < 16; d <<= 1) rs += __shfl_xor(rs, d, 32);
      lrun[i] = lrun[i] * alpha + rs;
      mrun[i] = mn;
      o0[i] *= alpha; o1[i] *= alpha; o2[i] *= alpha; o3[i] *= alpha;
      plds[w][i + 8 * hh][r] = (__bf16)v0;           // C-frag -> A-frag re-layout via LDS
      plds[w][i + 8 * hh][16 + r] = (__bf16)v1;
    }

    // ---- O += P V (P: 16x32 A frag; V tile in LDS: [key][dh]) ----
    const v16bf ap = load_frag_a(&plds[w][0][0], 32);  // same-wave DS, in-order
    o0 = wmma_bf16(ap, load_frag_b_strided(&Vlds[buf][0][0],  1, 64), o0);
    o1 = wmma_bf16(ap, load_frag_b_strided(&Vlds[buf][0][16], 1, 64), o1);
    o2 = wmma_bf16(ap, load_frag_b_strided(&Vlds[buf][0][32], 1, 64), o2);
    o3 = wmma_bf16(ap, load_frag_b_strided(&Vlds[buf][0][48], 1, 64), o3);

    __syncthreads();   // tile reads done before next stage overwrites a buffer
  }

  // ---- finalize: O / l, write attended [B,S,H,DH] as bf16 ----
#pragma unroll
  for (int i = 0; i < 8; ++i) {
    const float inv = 1.f / lrun[i];
    const size_t orow = ((size_t)(b * Sn + i0 + i + 8 * hh) * Hn + h) * DHn;
    attn[orow + r]      = (__bf16)(o0[i] * inv);
    attn[orow + 16 + r] = (__bf16)(o1[i] * inv);
    attn[orow + 32 + r] = (__bf16)(o2[i] * inv);
    attn[orow + 48 + r] = (__bf16)(o3[i] * inv);
  }
}

// ---------------- host side ----------------
extern "C" void kernel_launch(void* const* d_in, const int* in_sizes, int n_in,
                              void* d_out, int out_size, void* d_ws, size_t ws_size,
                              hipStream_t stream) {
  const float* tokens = (const float*)d_in[0];
  const float* abias  = (const float*)d_in[1];
  const float* meta   = (const float*)d_in[2];
  const unsigned char* kmask = (const unsigned char*)d_in[3];
  const float* ln1_g = (const float*)d_in[4];
  const float* ln1_b = (const float*)d_in[5];
  const float* Wq = (const float*)d_in[6];
  const float* bq = (const float*)d_in[7];
  const float* Wk = (const float*)d_in[8];
  const float* bk = (const float*)d_in[9];
  const float* Wv = (const float*)d_in[10];
  const float* bv = (const float*)d_in[11];
  const float* Wo = (const float*)d_in[12];
  const float* bo = (const float*)d_in[13];
  const float* Ws = (const float*)d_in[14];
  const float* bs = (const float*)d_in[15];
  const float* Wm = (const float*)d_in[16];
  const float* bm = (const float*)d_in[17];
  const float* ln2_g = (const float*)d_in[18];
  const float* ln2_b = (const float*)d_in[19];
  const float* W1 = (const float*)d_in[20];
  const float* b1 = (const float*)d_in[21];
  const float* W2 = (const float*)d_in[22];
  const float* b2 = (const float*)d_in[23];
  float* out = (float*)d_out;

  char* ws = (char*)d_ws;
  size_t off = 0;
  auto take = [&](size_t bytes) -> void* {
    void* p = ws + off;
    off += (bytes + 255) & ~(size_t)255;
    return p;
  };

  // activation region (32 MB) reused: [n|q|k|v] bf16 -> later ff bf16
  __bf16* n_bf  = (__bf16*)take((size_t)Mn * Dm * 2);
  __bf16* q_bf  = (__bf16*)take((size_t)Mn * Dm * 2);
  __bf16* k_bf  = (__bf16*)take((size_t)Mn * Dm * 2);
  __bf16* v_bf  = (__bf16*)take((size_t)Mn * Dm * 2);
  __bf16* ff_bf = n_bf;                       // aliases n|q|k|v after attention
  __bf16* attn_bf = (__bf16*)take((size_t)Mn * Dm * 2);
  __bf16* h_bf = attn_bf;                     // aliases attn after O-proj
  float*  x_f  = (float*)take((size_t)Mn * Dm * 4);
  __bf16* Wqt = (__bf16*)take((size_t)Dm * Dm * 2);
  __bf16* Wkt = (__bf16*)take((size_t)Dm * Dm * 2);
  __bf16* Wvt = (__bf16*)take((size_t)Dm * Dm * 2);
  __bf16* Wot = (__bf16*)take((size_t)Dm * Dm * 2);
  __bf16* W1t = (__bf16*)take((size_t)Dm * FFn * 2);
  __bf16* W2t = (__bf16*)take((size_t)FFn * Dm * 2);

  // weight convert+transpose (bf16, [N,K] layout so K is contiguous for B frags)
  {
    int g_dd = (Dm * Dm + 255) / 256;
    cvt_transpose_kernel<<<g_dd, 256, 0, stream>>>(Wq, Wqt, Dm, Dm);
    cvt_transpose_kernel<<<g_dd, 256, 0, stream>>>(Wk, Wkt, Dm, Dm);
    cvt_transpose_kernel<<<g_dd, 256, 0, stream>>>(Wv, Wvt, Dm, Dm);
    cvt_transpose_kernel<<<g_dd, 256, 0, stream>>>(Wo, Wot, Dm, Dm);
    int g_df = (Dm * FFn + 255) / 256;
    cvt_transpose_kernel<<<g_df, 256, 0, stream>>>(W1, W1t, Dm, FFn);  // -> [FF, D]
    cvt_transpose_kernel<<<g_df, 256, 0, stream>>>(W2, W2t, FFn, Dm);  // -> [D, FF]
  }

  // LN1 -> bf16
  ln_bf16_kernel<<<Mn, 256, 0, stream>>>(tokens, ln1_g, ln1_b, n_bf, Dm);

  // Q/K/V projections (bf16 out)
  dim3 gQKV(Dm / 64, Mn / 128);
  gemm_bf16_kernel<<<gQKV, 256, 0, stream>>>(n_bf, Wqt, bq, nullptr, nullptr, q_bf, Mn, Dm, Dm, 0);
  gemm_bf16_kernel<<<gQKV, 256, 0, stream>>>(n_bf, Wkt, bk, nullptr, nullptr, k_bf, Mn, Dm, Dm, 0);
  gemm_bf16_kernel<<<gQKV, 256, 0, stream>>>(n_bf, Wvt, bv, nullptr, nullptr, v_bf, Mn, Dm, Dm, 0);

  // flash attention with relation biases
  attention_kernel<<<dim3(Sn / 128, Hn, Bn), 256, 0, stream>>>(
      q_bf, k_bf, v_bf, abias, meta, kmask, Ws, bs, Wm, bm, attn_bf);

  // x = tokens + attn @ Wo + bo  (fp32)
  gemm_bf16_kernel<<<gQKV, 256, 0, stream>>>(attn_bf, Wot, bo, tokens, x_f, nullptr, Mn, Dm, Dm, 0);

  // LN2 -> bf16
  ln_bf16_kernel<<<Mn, 256, 0, stream>>>(x_f, ln2_g, ln2_b, h_bf, Dm);

  // ff = gelu(h @ W1 + b1)  (bf16)
  gemm_bf16_kernel<<<dim3(FFn / 64, Mn / 128), 256, 0, stream>>>(
      h_bf, W1t, b1, nullptr, nullptr, ff_bf, Mn, FFn, Dm, 1);

  // out = x + ff @ W2 + b2  (fp32)
  gemm_bf16_kernel<<<gQKV, 256, 0, stream>>>(ff_bf, W2t, b2, x_f, out, nullptr, Mn, Dm, FFn, 0);
}
